// Baseline_72662256714591
// MI455X (gfx1250) — compile-verified, matrix operand order checked by reading, not run on hardware
//
#include <hip/hip_runtime.h>
#include <hip/hip_bf16.h>

typedef __attribute__((ext_vector_type(16))) _Float16 v16h;
typedef __attribute__((ext_vector_type(8)))  _Float16 v8h;
typedef __attribute__((ext_vector_type(8)))  float    v8f;

// ---------------- weight convert: W[Ci,Co] f32 -> Wt[Co,Ci] f16 ----------------
__global__ void k_wconv(const float* __restrict__ W, _Float16* __restrict__ Wt, int Ci, int Co) {
  int gid = blockIdx.x * 256 + threadIdx.x;
  if (gid >= Ci * Co) return;
  int k = gid / Co, n = gid % Co;
  Wt[(size_t)n * Ci + k] = (_Float16)W[gid];
}

// ---------------- KNN k=16 (self or bipartite), LDS-tiled ----------------
__global__ void k_knn16(const float* __restrict__ q, const float* __restrict__ db,
                        int* __restrict__ nbr, int M, int N) {
  int b = blockIdx.y;
  int i = blockIdx.x * 128 + threadIdx.x;
  bool valid = i < M;
  float qx = 0.f, qy = 0.f, qz = 0.f;
  if (valid) {
    const float* qp = q + ((size_t)b * M + i) * 3;
    qx = qp[0]; qy = qp[1]; qz = qp[2];
  }
  float bd[16]; int bi[16];
#pragma unroll
  for (int k = 0; k < 16; ++k) { bd[k] = 3.0e38f; bi[k] = 0; }
  __shared__ float sp[128 * 3];
  for (int t0 = 0; t0 < N; t0 += 128) {
    int cnt = (N - t0) < 128 ? (N - t0) : 128;
    __syncthreads();
    for (int j = threadIdx.x; j < cnt * 3; j += 128)
      sp[j] = db[((size_t)b * N + t0) * 3 + j];
    __syncthreads();
    if (valid) {
      for (int j = 0; j < cnt; ++j) {
        float dx = sp[j * 3 + 0] - qx, dy = sp[j * 3 + 1] - qy, dz = sp[j * 3 + 2] - qz;
        float d = dx * dx + dy * dy + dz * dz;
        if (d < bd[15]) {
          bd[15] = d; bi[15] = t0 + j;
#pragma unroll
          for (int s = 15; s > 0; --s) {
            if (bd[s] < bd[s - 1]) {
              float td = bd[s]; bd[s] = bd[s - 1]; bd[s - 1] = td;
              int   ti = bi[s]; bi[s] = bi[s - 1]; bi[s - 1] = ti;
            }
          }
        }
      }
    }
  }
  if (valid) {
    int* np = nbr + ((size_t)b * M + i) * 16;
#pragma unroll
    for (int k = 0; k < 16; ++k) np[k] = bi[k];
  }
}

// ---------------- farthest point sampling, one block per batch ----------------
__global__ void k_fps(const float* __restrict__ pos, int* __restrict__ out,
                      float* __restrict__ dsc, int N, int M) {
  int b = blockIdx.x, tid = threadIdx.x;
  __shared__ float sv[256];
  __shared__ int   si[256];
  __shared__ int   sLast;
  const float* P = pos + (size_t)b * N * 3;
  float* D = dsc + (size_t)b * N;
  for (int n = tid; n < N; n += 256) D[n] = 1e10f;
  if (tid == 0) { out[(size_t)b * M] = 0; sLast = 0; }
  __syncthreads();
  for (int step = 1; step < M; ++step) {
    int last = sLast;
    float lx = P[last * 3 + 0], ly = P[last * 3 + 1], lz = P[last * 3 + 2];
    float bv = -1.f; int bidx = 0x7fffffff;
    for (int n = tid; n < N; n += 256) {
      float dx = P[n * 3 + 0] - lx, dy = P[n * 3 + 1] - ly, dz = P[n * 3 + 2] - lz;
      float d = dx * dx + dy * dy + dz * dz;
      float nd = fminf(D[n], d);
      D[n] = nd;
      if (nd > bv) { bv = nd; bidx = n; }
    }
    sv[tid] = bv; si[tid] = bidx;
    __syncthreads();
    for (int st = 128; st > 0; st >>= 1) {
      if (tid < st) {
        if (sv[tid + st] > sv[tid] || (sv[tid + st] == sv[tid] && si[tid + st] < si[tid])) {
          sv[tid] = sv[tid + st]; si[tid] = si[tid + st];
        }
      }
      __syncthreads();
    }
    if (tid == 0) { out[(size_t)b * M + step] = si[0]; sLast = si[0]; }
    __syncthreads();
  }
}

__global__ void k_gatherpos(const float* __restrict__ src, const int* __restrict__ idx,
                            float* __restrict__ dst, int M, int N) {
  int gid = blockIdx.x * 256 + threadIdx.x;
  if (gid >= 2 * M) return;
  int b = gid / M;
  int s = idx[gid];
  size_t so = ((size_t)b * N + s) * 3, dofs = (size_t)gid * 3;
  dst[dofs + 0] = src[so + 0]; dst[dofs + 1] = src[so + 1]; dst[dofs + 2] = src[so + 2];
}

// ---------------- ds-conv aggregate: spatial MLP + depthwise mean ----------------
__global__ void k_agg(const float* __restrict__ x, const float* __restrict__ ps,
                      const float* __restrict__ pd, const int* __restrict__ nbr,
                      const float* __restrict__ spW, const float* __restrict__ spb,
                      float* __restrict__ out, int N, int M, int Ci) {
  int p = blockIdx.x;           // 0 .. 2*M-1
  int b = p / M;
  __shared__ int   sIdx[16];
  __shared__ float rel[16][3];
  int tid = threadIdx.x;
  if (tid < 16) {
    int n = nbr[(size_t)p * 16 + tid];
    int g = b * N + n;
    sIdx[tid] = g;
    rel[tid][0] = ps[(size_t)g * 3 + 0] - pd[(size_t)p * 3 + 0];
    rel[tid][1] = ps[(size_t)g * 3 + 1] - pd[(size_t)p * 3 + 1];
    rel[tid][2] = ps[(size_t)g * 3 + 2] - pd[(size_t)p * 3 + 2];
  }
  __syncthreads();
  for (int c = tid; c < Ci; c += 64) {
    float w0 = spW[c], w1 = spW[Ci + c], w2 = spW[2 * Ci + c], bb = spb[c];
    float acc = 0.f;
#pragma unroll
    for (int j = 0; j < 16; ++j) {
      float w = rel[j][0] * w0 + rel[j][1] * w1 + rel[j][2] * w2 + bb;
      w = w > 0.f ? w : 0.f;
      acc += w * x[(size_t)sIdx[j] * Ci + c];
    }
    out[(size_t)p * Ci + c] = acc * 0.0625f;
  }
}

// ---------------- GEMM via v_wmma_f32_16x16x32_f16, one wave = 16x16 tile ----------------
__global__ __launch_bounds__(32) void k_gemm(const float* __restrict__ A,
                                             const _Float16* __restrict__ Wt,
                                             float* __restrict__ Y, int R, int Ci, int Co) {
  int lane = threadIdx.x;
  int row0 = blockIdx.x * 16, col0 = blockIdx.y * 16;
  int l15 = lane & 15, half = lane >> 4;
  int mr = row0 + l15, nc = col0 + l15;
  int koff = half * 8;
  v8f acc = {};
  for (int kt = 0; kt < Ci; kt += 32) {
    const float* ap = A + (size_t)mr * Ci + kt + koff;
    float4 a0 = *(const float4*)(ap + 0);
    float4 a1 = *(const float4*)(ap + 4);
    float4 a2 = *(const float4*)(ap + 16);
    float4 a3 = *(const float4*)(ap + 20);
    v16h av;
    av[0] = (_Float16)a0.x; av[1] = (_Float16)a0.y; av[2]  = (_Float16)a0.z; av[3]  = (_Float16)a0.w;
    av[4] = (_Float16)a1.x; av[5] = (_Float16)a1.y; av[6]  = (_Float16)a1.z; av[7]  = (_Float16)a1.w;
    av[8] = (_Float16)a2.x; av[9] = (_Float16)a2.y; av[10] = (_Float16)a2.z; av[11] = (_Float16)a2.w;
    av[12] = (_Float16)a3.x; av[13] = (_Float16)a3.y; av[14] = (_Float16)a3.z; av[15] = (_Float16)a3.w;
    const _Float16* bp = Wt + (size_t)nc * Ci + kt + koff;
    v8h b0 = *(const v8h*)(bp);
    v8h b1 = *(const v8h*)(bp + 16);
    v16h bv;
#pragma unroll
    for (int j = 0; j < 8; ++j) { bv[j] = b0[j]; bv[8 + j] = b1[j]; }
    acc = __builtin_amdgcn_wmma_f32_16x16x32_f16(false, av, false, bv, (short)0, acc, false, false);
  }
  int rbase = row0 + half * 8;
#pragma unroll
  for (int v = 0; v < 8; ++v)
    Y[(size_t)(rbase + v) * Co + nc] = acc[v];
}

// ---------------- batchnorm: per-channel stats then apply+activation ----------------
__global__ void k_bnstats(const float* __restrict__ Y, float* __restrict__ stats, int R, int C) {
  int c = blockIdx.x, tid = threadIdx.x;
  __shared__ float ss[256], sq[256];
  float s = 0.f, q = 0.f;
  for (int r = tid; r < R; r += 256) {
    float v = Y[(size_t)r * C + c];
    s += v; q += v * v;
  }
  ss[tid] = s; sq[tid] = q;
  __syncthreads();
  for (int st = 128; st > 0; st >>= 1) {
    if (tid < st) { ss[tid] += ss[tid + st]; sq[tid] += sq[tid + st]; }
    __syncthreads();
  }
  if (tid == 0) {
    float mu = ss[0] / (float)R;
    float var = sq[0] / (float)R - mu * mu;
    stats[c] = mu;
    stats[512 + c] = rsqrtf(var + 1e-5f);
  }
}

__global__ void k_bnapply(const float* __restrict__ Y, const float* __restrict__ stats,
                          const float* __restrict__ g, const float* __restrict__ be,
                          float* __restrict__ out, int R, int C, float slope) {
  int gid = blockIdx.x * 256 + threadIdx.x;
  if (gid >= R * C) return;
  int c = gid % C;
  float v = (Y[gid] - stats[c]) * stats[512 + c] * g[c] + be[c];
  out[gid] = v > 0.f ? v : slope * v;
}

// ---------------- interp: per-dst 3NN + weights, then weighted gather ----------------
__global__ void k_knn3w(const float* __restrict__ q, const float* __restrict__ db,
                        int* __restrict__ idx3, float* __restrict__ w3, int M, int N) {
  int gid = blockIdx.x * 128 + threadIdx.x;
  if (gid >= 2 * M) return;
  int b = gid / M;
  float qx = q[(size_t)gid * 3 + 0], qy = q[(size_t)gid * 3 + 1], qz = q[(size_t)gid * 3 + 2];
  float bd0 = 3.0e38f, bd1 = 3.0e38f, bd2 = 3.0e38f;
  int bi0 = 0, bi1 = 0, bi2 = 0;
  const float* P = db + (size_t)b * N * 3;
  for (int n = 0; n < N; ++n) {
    float dx = P[n * 3 + 0] - qx, dy = P[n * 3 + 1] - qy, dz = P[n * 3 + 2] - qz;
    float d = dx * dx + dy * dy + dz * dz;
    if (d < bd2) {
      if (d < bd0)      { bd2 = bd1; bi2 = bi1; bd1 = bd0; bi1 = bi0; bd0 = d; bi0 = n; }
      else if (d < bd1) { bd2 = bd1; bi2 = bi1; bd1 = d; bi1 = n; }
      else              { bd2 = d; bi2 = n; }
    }
  }
  float w0 = 1.f / (bd0 + 1e-8f), w1 = 1.f / (bd1 + 1e-8f), w2 = 1.f / (bd2 + 1e-8f);
  float inv = 1.f / (w0 + w1 + w2);
  idx3[(size_t)gid * 3 + 0] = bi0; idx3[(size_t)gid * 3 + 1] = bi1; idx3[(size_t)gid * 3 + 2] = bi2;
  w3[(size_t)gid * 3 + 0] = w0 * inv; w3[(size_t)gid * 3 + 1] = w1 * inv; w3[(size_t)gid * 3 + 2] = w2 * inv;
}

__global__ void k_interp(const float* __restrict__ feat, const int* __restrict__ idx3,
                         const float* __restrict__ w3, float* __restrict__ out,
                         int M, int N, int C) {
  int gid = blockIdx.x * 256 + threadIdx.x;
  if (gid >= 2 * M * C) return;
  int p = gid / C, c = gid % C;
  int b = p / M;
  size_t base = (size_t)b * N;
  float acc = 0.f;
#pragma unroll
  for (int j = 0; j < 3; ++j)
    acc += w3[(size_t)p * 3 + j] * feat[(base + idx3[(size_t)p * 3 + j]) * C + c];
  out[gid] = acc;
}

__global__ void k_concat(const float* __restrict__ a, const float* __restrict__ bsrc,
                         float* __restrict__ out, int R, int Ca, int Cb) {
  int C = Ca + Cb;
  int gid = blockIdx.x * 256 + threadIdx.x;
  if (gid >= R * C) return;
  int r = gid / C, c = gid % C;
  out[gid] = (c < Ca) ? a[(size_t)r * Ca + c] : bsrc[(size_t)r * Cb + (c - Ca)];
}

// ======================= host orchestration =======================
extern "C" void kernel_launch(void* const* d_in, const int* in_sizes, int n_in,
                              void* d_out, int out_size, void* d_ws, size_t ws_size,
                              hipStream_t stream) {
  (void)out_size; (void)ws_size;
  if (n_in < 90) return;

  // ---- resolve parameter pointers (supports pytree-sorted or insertion order) ----
  const bool caseI = in_sizes[0] > 1000;   // x first (insertion) vs conv0.be (sorted)
  const float *pos, *x_in;
  const float *cSpW[10], *cSpB[10], *cPwW[10], *cG[10], *cBe[10];
  // lin order used below: lin4, fusion3, lin3, fusion2, lin2, fusion1, lin1
  const float *lW[7], *lG[7], *lBe[7];
  if (!caseI) {
    for (int i = 0; i < 10; ++i) {
      int b = i * 6;
      cBe[i]  = (const float*)d_in[b + 0];
      cG[i]   = (const float*)d_in[b + 1];
      cPwW[i] = (const float*)d_in[b + 2];
      cSpW[i] = (const float*)d_in[b + 4];
      cSpB[i] = (const float*)d_in[b + 5];
    }
    // sorted lins: fusion1=60, fusion2=64, fusion3=68, lin1=72, lin2=76, lin3=80, lin4=84; [W,b,be,g]
    int bases[7] = {84, 68, 80, 64, 76, 60, 72};
    for (int j = 0; j < 7; ++j) {
      lW[j]  = (const float*)d_in[bases[j] + 0];
      lBe[j] = (const float*)d_in[bases[j] + 2];
      lG[j]  = (const float*)d_in[bases[j] + 3];
    }
    pos  = (const float*)d_in[88];
    x_in = (const float*)d_in[89];
  } else {
    x_in = (const float*)d_in[0];
    pos  = (const float*)d_in[1];
    for (int i = 0; i < 10; ++i) {
      int b = 2 + i * 6;
      cSpW[i] = (const float*)d_in[b + 0];
      cSpB[i] = (const float*)d_in[b + 1];
      cPwW[i] = (const float*)d_in[b + 2];
      cG[i]   = (const float*)d_in[b + 4];
      cBe[i]  = (const float*)d_in[b + 5];
    }
    // insertion lins: lin4=62, fusion3=66, lin3=70, fusion2=74, lin2=78, fusion1=82, lin1=86; [W,b,g,be]
    int bases[7] = {62, 66, 70, 74, 78, 82, 86};
    for (int j = 0; j < 7; ++j) {
      lW[j]  = (const float*)d_in[bases[j] + 0];
      lG[j]  = (const float*)d_in[bases[j] + 2];
      lBe[j] = (const float*)d_in[bases[j] + 3];
    }
  }

  // ---- workspace arena ----
  char* ws = (char*)d_ws;
  size_t off = 0;
  auto allocF = [&](size_t n) -> float* {
    float* p = (float*)(ws + off); off = (off + n * 4 + 255) & ~size_t(255); return p; };
  auto allocI = [&](size_t n) -> int* {
    int* p = (int*)(ws + off); off = (off + n * 4 + 255) & ~size_t(255); return p; };
  auto allocH = [&](size_t n) -> _Float16* {
    _Float16* p = (_Float16*)(ws + off); off = (off + n * 2 + 255) & ~size_t(255); return p; };

  float* x0 = allocF(262144);
  float* x1 = allocF(131072);
  float* x2 = allocF(65536);
  float* x3 = allocF(32768);
  float* x4 = allocF(16384);
  float* p1 = allocF(6144);
  float* p2 = allocF(1536);
  float* p3 = allocF(384);
  float* p4 = allocF(96);
  float* s1 = allocF(524288);
  float* s2 = allocF(524288);
  float* s3 = allocF(524288);
  float* stats = allocF(1024);
  float* w3 = allocF(24576);
  float* dsc = allocF(8192);
  int* nbr  = allocI(131072);
  int* fpsi = allocI(2048);
  int* idx3 = allocI(24576);

  const int convCi[10] = {32, 32, 32, 64, 64, 128, 128, 256, 256, 512};
  const int convCo[10] = {32, 32, 64, 64, 128, 128, 256, 256, 512, 512};
  const int linCi[7]   = {512, 512, 256, 256, 128, 128, 64};
  const int linCo[7]   = {256, 256, 128, 128, 64, 64, 32};

  _Float16* wtConv[10];
  _Float16* wtLin[7];
  for (int i = 0; i < 10; ++i) {
    int ne = convCi[i] * convCo[i];
    wtConv[i] = allocH((size_t)ne);
    k_wconv<<<(ne + 255) / 256, 256, 0, stream>>>(cPwW[i], wtConv[i], convCi[i], convCo[i]);
  }
  for (int j = 0; j < 7; ++j) {
    int ne = linCi[j] * linCo[j];
    wtLin[j] = allocH((size_t)ne);
    k_wconv<<<(ne + 255) / 256, 256, 0, stream>>>(lW[j], wtLin[j], linCi[j], linCo[j]);
  }

  // ---- helpers ----
  auto gemm = [&](const float* Aa, const _Float16* Wt, float* Yy, int R, int Ci, int Co) {
    dim3 g(R / 16, Co / 16);
    k_gemm<<<g, 32, 0, stream>>>(Aa, Wt, Yy, R, Ci, Co);
  };
  auto bn = [&](const float* Yy, const float* gg, const float* be, float* Oo,
                int R, int C, float slope) {
    k_bnstats<<<C, 256, 0, stream>>>(Yy, stats, R, C);
    int tot = R * C;
    k_bnapply<<<(tot + 255) / 256, 256, 0, stream>>>(Yy, stats, gg, be, Oo, R, C, slope);
  };
  auto knn = [&](const float* qq, const float* dd, int M, int N) {
    dim3 g((M + 127) / 128, 2);
    k_knn16<<<g, 128, 0, stream>>>(qq, dd, nbr, M, N);
  };
  auto agg = [&](const float* xf, const float* ps, const float* pd, int cv,
                 float* Oo, int N, int M, int Ci) {
    k_agg<<<2 * M, 64, 0, stream>>>(xf, ps, pd, nbr, cSpW[cv], cSpB[cv], Oo, N, M, Ci);
  };

  const float* Pl[5] = {pos, p1, p2, p3, p4};
  float* Xl[5] = {x0, x1, x2, x3, x4};
  const int Nl[5] = {4096, 1024, 256, 64, 16};

  // ---- level 0: two self-convs ----
  knn(pos, pos, 4096, 4096);
  agg(x_in, pos, pos, 0, s1, 4096, 4096, 32);
  gemm(s1, wtConv[0], s2, 8192, 32, 32);
  bn(s2, cG[0], cBe[0], s3, 8192, 32, 0.f);
  agg(s3, pos, pos, 1, s1, 4096, 4096, 32);
  gemm(s1, wtConv[1], s2, 8192, 32, 32);
  bn(s2, cG[1], cBe[1], x0, 8192, 32, 0.f);

  // ---- encoder levels 1..4 ----
  const float* xPrev = x0;
  int CiPrev = 32;
  for (int l = 1; l <= 4; ++l) {
    int Np = Nl[l - 1], Nc = Nl[l];
    const float* Pp = Pl[l - 1];
    float* Pc = (float*)Pl[l];
    int ca = 2 * l, cb = 2 * l + 1;
    int CoA = convCo[ca], CoB = convCo[cb];
    k_fps<<<2, 256, 0, stream>>>(Pp, fpsi, dsc, Np, Nc);
    k_gatherpos<<<(2 * Nc + 255) / 256, 256, 0, stream>>>(Pp, fpsi, Pc, Nc, Np);
    knn(Pc, Pp, Nc, Np);                                   // bipartite
    agg(xPrev, Pp, Pc, ca, s1, Np, Nc, CiPrev);
    gemm(s1, wtConv[ca], s2, 2 * Nc, CiPrev, CoA);
    bn(s2, cG[ca], cBe[ca], s3, 2 * Nc, CoA, 0.f);
    knn(Pc, Pc, Nc, Nc);                                   // self
    agg(s3, Pc, Pc, cb, s1, Nc, Nc, CoA);
    gemm(s1, wtConv[cb], s2, 2 * Nc, CoA, CoB);
    bn(s2, cG[cb], cBe[cb], Xl[l], 2 * Nc, CoB, 0.f);
    xPrev = Xl[l];
    CiPrev = CoB;
  }

  // ---- decoder: (lin4,fusion3) 4->3, (lin3,fusion2) 3->2, (lin2,fusion1) 2->1 ----
  const float* featSrc = x4;
  int curC = 512;
  for (int l = 4; l >= 2; --l) {
    int srcN = Nl[l], dstN = Nl[l - 1];
    int jLin = (4 - l) * 2, jFus = jLin + 1;
    int BM = 2 * dstN;
    k_knn3w<<<(BM + 127) / 128, 128, 0, stream>>>(Pl[l - 1], Pl[l], idx3, w3, dstN, srcN);
    k_interp<<<(BM * curC + 255) / 256, 256, 0, stream>>>(featSrc, idx3, w3, s1, dstN, srcN, curC);
    int Co1 = linCo[jLin];
    gemm(s1, wtLin[jLin], s2, BM, curC, Co1);
    bn(s2, lG[jLin], lBe[jLin], s3, BM, Co1, 0.01f);
    int Cskip = convCo[2 * (l - 1) + 1];
    k_concat<<<(BM * (Co1 + Cskip) + 255) / 256, 256, 0, stream>>>(s3, Xl[l - 1], s1, BM, Co1, Cskip);
    int CoF = linCo[jFus];
    gemm(s1, wtLin[jFus], s2, BM, Co1 + Cskip, CoF);
    bn(s2, lG[jFus], lBe[jFus], s3, BM, CoF, 0.01f);
    featSrc = s3;
    curC = CoF;
  }

  // ---- final: interp level1 -> level0, lin1, concat with x0 into d_out ----
  k_knn3w<<<(8192 + 127) / 128, 128, 0, stream>>>(pos, Pl[1], idx3, w3, 4096, 1024);
  k_interp<<<(8192 * 64 + 255) / 256, 256, 0, stream>>>(featSrc, idx3, w3, s1, 4096, 1024, 64);
  gemm(s1, wtLin[6], s2, 8192, 64, 32);
  bn(s2, lG[6], lBe[6], s3, 8192, 32, 0.01f);
  k_concat<<<(8192 * 64 + 255) / 256, 256, 0, stream>>>(s3, x0, (float*)d_out, 8192, 32, 32);
}
